// Attn_81819126989106
// MI455X (gfx1250) — matrix operand order repaired and checked by reading, not compiled, not run on hardware
//
#include <hip/hip_runtime.h>
#include <hip/hip_bf16.h>
#include <math.h>

// Shapes
#define Bb 32
#define Ss 2048
#define Hh 1024
#define Mrows (Bb * Ss)   // 65536

typedef __bf16 bf16_t;
typedef __attribute__((ext_vector_type(8)))  __bf16 v8bf;
typedef __attribute__((ext_vector_type(16))) __bf16 v16bf;
typedef __attribute__((ext_vector_type(8)))  float  v8f;

#define APAD 8                      // LDS row padding (elements)
#define ASTRIDE (Hh + APAD)         // 1032 bf16 = 2064 B (16B aligned)
#define MT 64                       // M rows per workgroup (4 WMMA tiles)
#define NW 4                        // waves per workgroup
#define NTHREADS (NW * 32)

// Native TRANS32 tanh on gfx1250; guarded fallback to libm.
#if defined(__has_builtin)
#if __has_builtin(__builtin_amdgcn_tanhf)
#define FAST_TANH(x) __builtin_amdgcn_tanhf(x)
#endif
#endif
#ifndef FAST_TANH
#define FAST_TANH(x) tanhf(x)
#endif

// ---------------------------------------------------------------------------
// Kernel 0: convert W_enc (fp32, row-major [H,H]) -> bf16 in workspace.
// ---------------------------------------------------------------------------
__global__ __launch_bounds__(256) void wcvt_kernel(
    const float* __restrict__ W, bf16_t* __restrict__ Wb)
{
    const int i = (blockIdx.x * 256 + threadIdx.x) * 8;
    const float4 f0 = *(const float4*)(W + i);
    const float4 f1 = *(const float4*)(W + i + 4);
    v8bf t;
    t[0] = (bf16_t)f0.x; t[1] = (bf16_t)f0.y; t[2] = (bf16_t)f0.z; t[3] = (bf16_t)f0.w;
    t[4] = (bf16_t)f1.x; t[5] = (bf16_t)f1.y; t[6] = (bf16_t)f1.z; t[7] = (bf16_t)f1.w;
    *(v8bf*)(Wb + i) = t;
}

// ---------------------------------------------------------------------------
// Kernel 1: dec_feature[b, o] = decoder_hidden[b, :] . W_dec[o, :] + b_dec[o]
// ---------------------------------------------------------------------------
__global__ __launch_bounds__(256) void dec_feature_kernel(
    const float* __restrict__ dh,
    const float* __restrict__ Wdec,
    const float* __restrict__ bdec,
    float* __restrict__ dec_out)
{
    int b = blockIdx.x;
    int o = blockIdx.y * blockDim.x + threadIdx.x;
    const float* x = dh + (size_t)b * Hh;
    const float* w = Wdec + (size_t)o * Hh;
    float s = 0.0f;
    #pragma unroll 8
    for (int h = 0; h < Hh; ++h) s = fmaf(x[h], w[h], s);
    dec_out[b * Hh + o] = s + bdec[o];
}

// ---------------------------------------------------------------------------
// Kernel 2: fused  scores[r] = v . tanh(X[r,:] @ Wenc^T + dec[b,:] + cov[r]*wcov)
// 4-wave WG shares a 64-row A tile in LDS (staged once, bf16, tile-interleaved
// row order so all DS offsets are immediates). Waves split the 64 N-tiles;
// each B fragment feeds 4 WMMAs. The affine term (dec + cov*wcov) is folded
// into the WMMA C initializer. Deterministic LDS tree sum combines waves.
// ---------------------------------------------------------------------------
__global__ __launch_bounds__(NTHREADS) void scores_kernel(
    const float* __restrict__ X,       // encoder_outputs [M, H] fp32
    const bf16_t* __restrict__ Wb,     // [H, H] bf16 (workspace)
    const float* __restrict__ cov,     // [B, S]
    const float* __restrict__ dec,     // [B, H] (workspace)
    const float* __restrict__ wcov,    // [H]
    const float* __restrict__ vvec,    // [H]
    float* __restrict__ out)           // [M] scores (flat [B,1,S])
{
    __shared__ bf16_t Abuf[MT * ASTRIDE];   // 64 rows, ~129 KB
    __shared__ float  Red[NW][MT];          // per-wave row partials, 1 KB

    const int tid  = threadIdx.x;
    const int wave = tid >> 5;
    const int lane = tid & 31;
    const int half = lane >> 4;
    const int l15  = lane & 15;
    const int r0   = blockIdx.x * MT;
    const int b    = r0 >> 11;         // 64 | 2048: block shares batch b

    // ---- Stage A cooperatively: MT x 1024 fp32 -> bf16 LDS, once ----
    // Row m goes to slot (m&15)*4 + (m>>4): a lane's 4 tile rows are adjacent,
    // so every hot-loop ds_load offset fits the 16-bit immediate field.
    #pragma unroll 4
    for (int it = 0; it < (MT * Hh) / (NTHREADS * 8); ++it) {   // 64 iters
        const int flat = (it * NTHREADS + tid) * 8;
        const int m = flat >> 10;
        const int k = flat & (Hh - 1);
        const int slot = (m & 15) * 4 + (m >> 4);
        const float* src = X + (size_t)(r0 + m) * Hh + k;
        const float4 f0 = *(const float4*)(src);
        const float4 f1 = *(const float4*)(src + 4);
        v8bf t;
        t[0] = (bf16_t)f0.x; t[1] = (bf16_t)f0.y; t[2] = (bf16_t)f0.z; t[3] = (bf16_t)f0.w;
        t[4] = (bf16_t)f1.x; t[5] = (bf16_t)f1.y; t[6] = (bf16_t)f1.z; t[7] = (bf16_t)f1.w;
        *(v8bf*)&Abuf[slot * ASTRIDE + k] = t;
    }
    __syncthreads();

    // coverage for the 32 C-tile rows this lane accumulates (4 tiles x 8)
    const float* covb = cov + (size_t)b * Ss;
    const int s_base = r0 - b * Ss;
    float covj[32];
    #pragma unroll
    for (int t = 0; t < 4; ++t)
        #pragma unroll
        for (int j = 0; j < 8; ++j)
            covj[t * 8 + j] = covb[s_base + t * 16 + half * 8 + j];

    float part[32];
    #pragma unroll
    for (int i = 0; i < 32; ++i) part[i] = 0.f;

    const float* decb = dec + (size_t)b * Hh;

    // One LDS base for this lane; tile t lives at +t*ASTRIDE (immediate).
    const bf16_t* albase = &Abuf[(l15 * 4) * ASTRIDE] + half * 8;

    for (int i = 0; i < 16; ++i) {               // this wave's 16 N tiles
        const int nt = wave * 16 + i;
        const int n  = nt * 16 + l15;
        const bf16_t* wptr = Wb + (size_t)n * Hh + half * 8;

        const float dn  = decb[n];
        const float wcn = wcov[n];
        const float vn  = vvec[n];

        // Fold the affine term (dec + cov*wcov) into the C initializer.
        v8f acc[4];
        #pragma unroll
        for (int t = 0; t < 4; ++t)
            #pragma unroll
            for (int j = 0; j < 8; ++j)
                acc[t][j] = fmaf(covj[t * 8 + j], wcn, dn);

        #pragma unroll 4
        for (int k0 = 0; k0 < Hh; k0 += 32) {    // 32 K steps
            // B fragment: two 16-byte chunks of W row n (k 8h..8h+7, 16+8h..) — shared by 4 WMMAs
            const v8bf blo = *(const v8bf*)(wptr + k0);
            const v8bf bhi = *(const v8bf*)(wptr + k0 + 16);
            const v16bf bf = __builtin_shufflevector(
                blo, bhi, 0, 1, 2, 3, 4, 5, 6, 7, 8, 9, 10, 11, 12, 13, 14, 15);
            #pragma unroll
            for (int t = 0; t < 4; ++t) {
                const v8bf alo = *(const v8bf*)(albase + t * ASTRIDE + k0);
                const v8bf ahi = *(const v8bf*)(albase + t * ASTRIDE + k0 + 16);
                const v16bf a = __builtin_shufflevector(
                    alo, ahi, 0, 1, 2, 3, 4, 5, 6, 7, 8, 9, 10, 11, 12, 13, 14, 15);
                acc[t] = __builtin_amdgcn_wmma_f32_16x16x32_bf16(
                    false, a, false, bf, (short)0, acc[t], false, false);
            }
        }

        // tanh + dot with v
        #pragma unroll
        for (int t = 0; t < 4; ++t)
            #pragma unroll
            for (int j = 0; j < 8; ++j)
                part[t * 8 + j] += FAST_TANH(acc[t][j]) * vn;
    }

    // Reduce each row's partial across the 16 lanes of its half
    #pragma unroll
    for (int i = 0; i < 32; ++i) {
        float p = part[i];
        p += __shfl_xor(p, 1, 32);
        p += __shfl_xor(p, 2, 32);
        p += __shfl_xor(p, 4, 32);
        p += __shfl_xor(p, 8, 32);
        part[i] = p;
    }
    // Deterministic cross-wave combine via LDS (no float atomics).
    if (l15 == 0) {
        #pragma unroll
        for (int t = 0; t < 4; ++t)
            #pragma unroll
            for (int j = 0; j < 8; ++j)
                Red[wave][t * 16 + half * 8 + j] = part[t * 8 + j];
    }
    __syncthreads();
    if (tid < MT) {
        out[r0 + tid] = (Red[0][tid] + Red[1][tid]) + (Red[2][tid] + Red[3][tid]);
    }
}

// ---------------------------------------------------------------------------
// Kernel 3: in-place softmax over S=2048 per batch row of d_out.
// ---------------------------------------------------------------------------
__global__ __launch_bounds__(256) void softmax_kernel(float* __restrict__ out)
{
    __shared__ float red[256];
    const int b = blockIdx.x;
    const int t = threadIdx.x;
    float* row = out + (size_t)b * Ss;

    float vals[8];
    float m = -INFINITY;
    #pragma unroll
    for (int i = 0; i < 8; ++i) {
        vals[i] = row[t + i * 256];
        m = fmaxf(m, vals[i]);
    }
    red[t] = m;
    __syncthreads();
    for (int s = 128; s > 0; s >>= 1) {
        if (t < s) red[t] = fmaxf(red[t], red[t + s]);
        __syncthreads();
    }
    m = red[0];
    __syncthreads();

    float sum = 0.f;
    #pragma unroll
    for (int i = 0; i < 8; ++i) {
        vals[i] = __expf(vals[i] - m);
        sum += vals[i];
    }
    red[t] = sum;
    __syncthreads();
    for (int s = 128; s > 0; s >>= 1) {
        if (t < s) red[t] += red[t + s];
        __syncthreads();
    }
    const float inv = 1.0f / red[0];
    #pragma unroll
    for (int i = 0; i < 8; ++i) row[t + i * 256] = vals[i] * inv;
}

// ---------------------------------------------------------------------------
extern "C" void kernel_launch(void* const* d_in, const int* in_sizes, int n_in,
                              void* d_out, int out_size, void* d_ws, size_t ws_size,
                              hipStream_t stream)
{
    const float* dh   = (const float*)d_in[0]; // decoder_hidden [32,1,1024]
    const float* enc  = (const float*)d_in[1]; // encoder_outputs [32,2048,1024]
    const float* cov  = (const float*)d_in[2]; // coverage [32,2048]
    const float* Wenc = (const float*)d_in[3]; // [1024,1024]
    const float* Wdec = (const float*)d_in[4]; // [1024,1024]
    const float* bdec = (const float*)d_in[5]; // [1024]
    const float* wcov = (const float*)d_in[6]; // [1024]
    const float* vv   = (const float*)d_in[7]; // [1024]
    float* out = (float*)d_out;                // [32,1,2048] flat

    // Workspace: [0, 2MB) = W_enc bf16, then dec_feature [32,1024] fp32
    bf16_t* Wb  = (bf16_t*)d_ws;
    float*  dec = (float*)((char*)d_ws + (size_t)Hh * Hh * sizeof(bf16_t));

    wcvt_kernel<<<(Hh * Hh) / (256 * 8), 256, 0, stream>>>(Wenc, Wb);
    dec_feature_kernel<<<dim3(Bb, Hh / 256), 256, 0, stream>>>(dh, Wdec, bdec, dec);
    scores_kernel<<<Mrows / MT, NTHREADS, 0, stream>>>(enc, Wb, cov, dec, wcov, vv, out);
    softmax_kernel<<<Bb, 256, 0, stream>>>(out);
}
